// AttentionBlock_9706626089928
// MI455X (gfx1250) — matrix operand order repaired and checked by reading, not compiled, not run on hardware
//
#include <hip/hip_runtime.h>
#include <hip/hip_bf16.h>
#include <cstdint>

// ---------------------------------------------------------------------------
// Transformer block for gfx1250 (MI455X): all matmuls via v_wmma_f32_16x16x32_bf16
// Async global->LDS staging (ASYNCcnt) where the toolchain exposes it.
// ---------------------------------------------------------------------------

typedef __attribute__((ext_vector_type(16))) __bf16 v16bf;
typedef __attribute__((ext_vector_type(8)))  float  v8f;
typedef int v4i __attribute__((ext_vector_type(4)));

union Frag {
    v16bf    v;
    uint32_t u[8];
    uint4    q[2];
};

typedef __attribute__((address_space(1))) v4i v4i_g;   // global int4
typedef __attribute__((address_space(3))) v4i v4i_l;   // LDS int4

#if __has_builtin(__builtin_amdgcn_global_load_async_to_lds_b128)
#define ASYNC_CP16(dst, src)                                                   \
    __builtin_amdgcn_global_load_async_to_lds_b128(                            \
        (v4i_g*)(uintptr_t)(src),                                              \
        (v4i_l*)(uint32_t)(uintptr_t)(dst), 0, 0)
#if __has_builtin(__builtin_amdgcn_s_wait_asynccnt)
#define ASYNC_WAIT() __builtin_amdgcn_s_wait_asynccnt(0)
#else
#define ASYNC_WAIT() asm volatile("s_wait_asynccnt 0x0" ::: "memory")
#endif
#else
#define ASYNC_CP16(dst, src) (*(uint4*)(dst) = *(const uint4*)(src))
#define ASYNC_WAIT() ((void)0)
#endif

#define DIMD      1024
#define SEQ_L     2048
#define NHEAD     16
#define HDIM      64
#define MROWS     4096      // B * L
#define WPAD      66        // padded k-row (halves) for the transposed W tile

// ---------------------------------------------------------------------------
// LayerNorm: f32 [M, D] -> bf16 [M, D]
// ---------------------------------------------------------------------------
__global__ __launch_bounds__(256) void ln_kernel(
    const float* __restrict__ x, const float* __restrict__ w,
    const float* __restrict__ b, __bf16* __restrict__ out)
{
    const int row = blockIdx.x;
    const float* xr = x + (size_t)row * DIMD;

    float s = 0.f, s2 = 0.f;
    float vals[4];
#pragma unroll
    for (int i = 0; i < 4; ++i) {
        float v = xr[threadIdx.x + i * 256];
        vals[i] = v; s += v; s2 += v * v;
    }
#pragma unroll
    for (int off = 16; off > 0; off >>= 1) {
        s  += __shfl_xor(s,  off, 32);
        s2 += __shfl_xor(s2, off, 32);
    }
    __shared__ float ps[8], ps2[8];
    const int wid = threadIdx.x >> 5, lane = threadIdx.x & 31;
    if (lane == 0) { ps[wid] = s; ps2[wid] = s2; }
    __syncthreads();
    float ts = 0.f, ts2 = 0.f;
#pragma unroll
    for (int i = 0; i < 8; ++i) { ts += ps[i]; ts2 += ps2[i]; }

    const float mean = ts * (1.f / DIMD);
    const float var  = ts2 * (1.f / DIMD) - mean * mean;
    const float rstd = rsqrtf(var + 1e-5f);

    __bf16* orow = out + (size_t)row * DIMD;
#pragma unroll
    for (int i = 0; i < 4; ++i) {
        int d = threadIdx.x + i * 256;
        orow[d] = (__bf16)((vals[i] - mean) * rstd * w[d] + b[d]);
    }
}

// ---------------------------------------------------------------------------
// GEMM: out = act(A[M,K](bf16) @ W[K,N](f32->bf16) + bias) (+ resid)
// Block tile 128x128, BK=64, 256 threads (8 waves), wave tile 32x64.
// A tile: async global->LDS copy (ASYNCcnt). W tile: f32 regs -> packed bf16
// pairs in LDS. Double-buffered; one barrier per 64-K stage.
// ---------------------------------------------------------------------------
template <bool GELU, bool RES, bool OUTBF>
__global__ __launch_bounds__(256) void gemm_kernel(
    const __bf16* __restrict__ A, const float* __restrict__ W,
    const float* __restrict__ bias, const float* __restrict__ resid,
    void* __restrict__ outp, int K, int N)
{
    __shared__ __bf16 sA[2][128][64];     // [buf][m][k]
    __shared__ __bf16 sW[2][128][WPAD];   // [buf][n][k] (padded vs bank conflicts)

    const int tid  = threadIdx.x;
    const int lane = tid & 31, wid = tid >> 5;
    const int wm   = wid & 3,  wn  = wid >> 2;       // 4 x 2 wave grid
    const int bm   = blockIdx.y * 128, bn = blockIdx.x * 128;
    const int kh   = lane >> 4, nl = lane & 15;

    v8f acc[2][4];
    const v8f vz = {0.f, 0.f, 0.f, 0.f, 0.f, 0.f, 0.f, 0.f};
#pragma unroll
    for (int i = 0; i < 2; ++i)
#pragma unroll
        for (int j = 0; j < 4; ++j) acc[i][j] = vz;

    // staging geometry
    const int arow = tid >> 1, aseg = (tid & 1) * 32;   // A: 32 bf16 per thread
    const int wk   = (tid >> 3) * 2;                    // W: k-row pair
    const int wseg = (tid & 7) * 16;                    // W: 16 n per thread

    float4 rw0[4], rw1[4];

    auto stage_a = [&](int k0, int buf) {
        const __bf16* gp = A + (size_t)(bm + arow) * K + k0 + aseg;
        __bf16* lp = &sA[buf][arow][aseg];
#pragma unroll
        for (int i = 0; i < 4; ++i) ASYNC_CP16(lp + i * 8, gp + i * 8);
    };
    auto load_w = [&](int k0) {
        const float4* p0 = (const float4*)(W + (size_t)(k0 + wk) * N + bn + wseg);
        const float4* p1 = (const float4*)(W + (size_t)(k0 + wk + 1) * N + bn + wseg);
#pragma unroll
        for (int i = 0; i < 4; ++i) { rw0[i] = p0[i]; rw1[i] = p1[i]; }
        // prefetch the weight stream two stages ahead into GL2
        if (k0 + 128 < K)
            __builtin_prefetch(W + (size_t)(k0 + 128 + wk) * N + bn + wseg, 0, 1);
    };
    auto store_w = [&](int buf) {
        float f0[16] = {rw0[0].x, rw0[0].y, rw0[0].z, rw0[0].w,
                        rw0[1].x, rw0[1].y, rw0[1].z, rw0[1].w,
                        rw0[2].x, rw0[2].y, rw0[2].z, rw0[2].w,
                        rw0[3].x, rw0[3].y, rw0[3].z, rw0[3].w};
        float f1[16] = {rw1[0].x, rw1[0].y, rw1[0].z, rw1[0].w,
                        rw1[1].x, rw1[1].y, rw1[1].z, rw1[1].w,
                        rw1[2].x, rw1[2].y, rw1[2].z, rw1[2].w,
                        rw1[3].x, rw1[3].y, rw1[3].z, rw1[3].w};
#pragma unroll
        for (int ii = 0; ii < 16; ++ii) {
            union { __bf16 h[2]; uint32_t u; } pk;
            pk.h[0] = (__bf16)f0[ii];   // k = wk
            pk.h[1] = (__bf16)f1[ii];   // k = wk + 1
            *(uint32_t*)&sW[buf][wseg + ii][wk] = pk.u;
        }
    };

    stage_a(0, 0);
    load_w(0);
    store_w(0);
    ASYNC_WAIT();
    __syncthreads();

    const int nstages = K >> 6;
    for (int st = 0; st < nstages; ++st) {
        const int cur = st & 1;
        if (st + 1 < nstages) {
            stage_a((st + 1) << 6, cur ^ 1);   // async copies overlap compute
            load_w((st + 1) << 6);
        }

        // ---- compute: 2 k-substeps x (2x4) WMMA ----
#pragma unroll
        for (int ks = 0; ks < 2; ++ks) {
            Frag a[2], b[4];
#pragma unroll
            for (int sm = 0; sm < 2; ++sm) {
                const uint32_t* rp =
                    (const uint32_t*)&sA[cur][wm * 32 + sm * 16 + nl][0];
#pragma unroll
                for (int j = 0; j < 4; ++j) {
                    a[sm].u[j]     = rp[ks * 16 + kh * 4 + j];
                    a[sm].u[4 + j] = rp[ks * 16 + 8 + kh * 4 + j];
                }
            }
#pragma unroll
            for (int sn = 0; sn < 4; ++sn) {
                const uint32_t* rp =
                    (const uint32_t*)&sW[cur][wn * 64 + sn * 16 + nl][0];
#pragma unroll
                for (int j = 0; j < 8; ++j) b[sn].u[j] = rp[ks * 16 + (j >> 2) * 8 + kh * 4 + (j & 3)];
            }
#pragma unroll
            for (int sm = 0; sm < 2; ++sm)
#pragma unroll
                for (int sn = 0; sn < 4; ++sn)
                    acc[sm][sn] = __builtin_amdgcn_wmma_f32_16x16x32_bf16(
                        false, a[sm].v, false, b[sn].v, (short)0, acc[sm][sn],
                        false, false);
        }

        if (st + 1 < nstages) store_w(cur ^ 1);
        ASYNC_WAIT();
        __syncthreads();
    }

    // ---- epilogue ----
#pragma unroll
    for (int sm = 0; sm < 2; ++sm)
#pragma unroll
        for (int sn = 0; sn < 4; ++sn) {
            const int col = bn + wn * 64 + sn * 16 + nl;
            const float bv = bias[col];
#pragma unroll
            for (int v = 0; v < 8; ++v) {
                const int row = bm + wm * 32 + sm * 16 + v + 8 * kh;
                float val = acc[sm][sn][v] + bv;
                if (GELU) val = 0.5f * val * (1.f + erff(val * 0.70710678f));
                if (RES)  val += resid[(size_t)row * N + col];
                if (OUTBF) ((__bf16*)outp)[(size_t)row * N + col] = (__bf16)val;
                else       ((float*)outp)[(size_t)row * N + col]  = val;
            }
        }
}

// ---------------------------------------------------------------------------
// Flash attention (causal). qkv bf16 [M, 3*D] (q | k | v per head).
// Block: 64 queries for one (batch, head); 4 waves x 16 queries.
// S^T = K x Q^T, online softmax over the M (key) dim, O^T = V^T x P^T.
// K tile staged via async global->LDS; fragments batched 4-wide per WMMA chain.
// ---------------------------------------------------------------------------
__global__ __launch_bounds__(128) void attn_kernel(
    const __bf16* __restrict__ qkv, __bf16* __restrict__ out)
{
    __shared__ __bf16 sK[64][64];        // [key][dim]
    __shared__ __bf16 sVT[64][64];       // [dim][key]
    __shared__ __bf16 sPT[4][16][64];    // per-wave [q][key] bf16 probs

    const int tid = threadIdx.x, lane = tid & 31, wid = tid >> 5;
    const int qt = blockIdx.x;
    const int bh = blockIdx.y, bidx = bh >> 4, h = bh & 15;
    const int D3 = 3 * DIMD;
    const int kh = lane >> 4, nl = lane & 15;

    const size_t seqbase = (size_t)bidx * SEQ_L;
    const int    qpos    = qt * 64 + wid * 16 + nl;     // position in sequence
    const size_t qrow    = seqbase + qpos;

    // Q as B-fragments (Q^T), dims split into two k-chunks of 32
    Frag bq[2];
#pragma unroll
    for (int c = 0; c < 2; ++c) {
        const uint4* src =
            (const uint4*)(qkv + qrow * D3 + h * HDIM + c * 32 + kh * 8);
        bq[c].q[0] = src[0];
        bq[c].q[1] = src[2];
    }

    float m = -INFINITY, lsum = 0.f;
    v8f o[4];
    const v8f vz = {0.f, 0.f, 0.f, 0.f, 0.f, 0.f, 0.f, 0.f};
#pragma unroll
    for (int f = 0; f < 4; ++f) o[f] = vz;

    const int key0 = tid >> 1, dseg = (tid & 1) * 32;
    const int ntiles = qt + 1;

    for (int j = 0; j < ntiles; ++j) {
        __syncthreads();
        const size_t krow = seqbase + j * 64 + key0;
        // stage K row-major via async copies
        const __bf16* kgp = qkv + krow * D3 + DIMD + h * HDIM + dseg;
        __bf16* klp = &sK[key0][dseg];
#pragma unroll
        for (int i = 0; i < 4; ++i) ASYNC_CP16(klp + i * 8, kgp + i * 8);
        // stage V transposed (through registers)
        const uint4* vsrc =
            (const uint4*)(qkv + krow * D3 + 2 * DIMD + h * HDIM + dseg);
        union { uint4 q[4]; __bf16 hv[32]; } vb;
        vb.q[0] = vsrc[0]; vb.q[1] = vsrc[1]; vb.q[2] = vsrc[2]; vb.q[3] = vsrc[3];
#pragma unroll
        for (int d = 0; d < 32; ++d) sVT[dseg + d][key0] = vb.hv[d];
        ASYNC_WAIT();
        __syncthreads();

        // ---- scores S^T (4 key-subtiles of 16), fragment loads batched ----
        v8f sc[4];
#pragma unroll
        for (int s = 0; s < 4; ++s) sc[s] = vz;
#pragma unroll
        for (int c = 0; c < 2; ++c) {
            Frag ak[4];
#pragma unroll
            for (int s = 0; s < 4; ++s) {
                const uint32_t* rp = (const uint32_t*)&sK[s * 16 + nl][0];
#pragma unroll
                for (int jj = 0; jj < 4; ++jj) {
                    ak[s].u[jj]     = rp[c * 16 + kh * 4 + jj];
                    ak[s].u[4 + jj] = rp[c * 16 + 8 + kh * 4 + jj];
                }
            }
#pragma unroll
            for (int s = 0; s < 4; ++s)
                sc[s] = __builtin_amdgcn_wmma_f32_16x16x32_bf16(
                    false, ak[s].v, false, bq[c].v, (short)0, sc[s], false, false);
        }

        // ---- causal mask + scale + online softmax ----
        float tm = -INFINITY;
#pragma unroll
        for (int s = 0; s < 4; ++s)
#pragma unroll
            for (int v = 0; v < 8; ++v) {
                const int kpos = j * 64 + s * 16 + v + 8 * kh;
                float sv = sc[s][v] * 0.125f;          // SCALE^2 = 1/sqrt(64)
                if (kpos > qpos) sv = -INFINITY;
                sc[s][v] = sv;
                tm = fmaxf(tm, sv);
            }
        tm = fmaxf(tm, __shfl_xor(tm, 16, 32));
        const float mn    = fmaxf(m, tm);
        const float mexp  = (mn == -INFINITY) ? 0.f : mn;
        const float alpha = __expf(m - mexp);

        float psum = 0.f;
#pragma unroll
        for (int s = 0; s < 4; ++s)
#pragma unroll
            for (int v = 0; v < 8; ++v) {
                const float p = __expf(sc[s][v] - mexp);
                psum += p;
                sPT[wid][nl][s * 16 + v + 8 * kh] = (__bf16)p;
            }
        psum += __shfl_xor(psum, 16, 32);
        lsum = lsum * alpha + psum;
        m = mn;
#pragma unroll
        for (int f = 0; f < 4; ++f)
#pragma unroll
            for (int v = 0; v < 8; ++v) o[f][v] = o[f][v] * alpha;

        // ---- O^T += V^T x P^T (fragment loads batched) ----
#pragma unroll
        for (int c = 0; c < 2; ++c) {
            Frag bp;
            const uint32_t* pp = (const uint32_t*)&sPT[wid][nl][0];
#pragma unroll
            for (int jj = 0; jj < 8; ++jj) bp.u[jj] = pp[c * 16 + (jj >> 2) * 8 + kh * 4 + (jj & 3)];
            Frag av[4];
#pragma unroll
            for (int ds = 0; ds < 4; ++ds) {
                const uint32_t* rp = (const uint32_t*)&sVT[ds * 16 + nl][0];
#pragma unroll
                for (int jj = 0; jj < 4; ++jj) {
                    av[ds].u[jj]     = rp[c * 16 + kh * 4 + jj];
                    av[ds].u[4 + jj] = rp[c * 16 + 8 + kh * 4 + jj];
                }
            }
#pragma unroll
            for (int ds = 0; ds < 4; ++ds)
                o[ds] = __builtin_amdgcn_wmma_f32_16x16x32_bf16(
                    false, av[ds].v, false, bp.v, (short)0, o[ds], false, false);
        }
    }

    // ---- normalize + write O (bf16, one 16B store per fragment) ----
    const float inv = 1.f / lsum;
#pragma unroll
    for (int ds = 0; ds < 4; ++ds) {
        union { __bf16 hv[8]; uint4 q; } ob;
#pragma unroll
        for (int v = 0; v < 8; ++v) ob.hv[v] = (__bf16)(o[ds][v] * inv);
        __bf16* dst = out + qrow * DIMD + h * HDIM + ds * 16 + kh * 8;
        *(uint4*)dst = ob.q;
    }
}

// ---------------------------------------------------------------------------
// Host-side orchestration
// ---------------------------------------------------------------------------
extern "C" void kernel_launch(void* const* d_in, const int* in_sizes, int n_in,
                              void* d_out, int out_size, void* d_ws,
                              size_t ws_size, hipStream_t stream)
{
    (void)in_sizes; (void)n_in; (void)out_size; (void)ws_size;
    const float* x      = (const float*)d_in[0];
    // d_in[1] = mask (unused; causal mask applied analytically)
    const float* ln1_w  = (const float*)d_in[2];
    const float* ln1_b  = (const float*)d_in[3];
    const float* qkv_w  = (const float*)d_in[4];
    const float* qkv_b  = (const float*)d_in[5];
    const float* proj_w = (const float*)d_in[6];
    const float* proj_b = (const float*)d_in[7];
    const float* ln2_w  = (const float*)d_in[8];
    const float* ln2_b  = (const float*)d_in[9];
    const float* ffn_w1 = (const float*)d_in[10];
    const float* ffn_b1 = (const float*)d_in[11];
    const float* ffn_w2 = (const float*)d_in[12];
    const float* ffn_b2 = (const float*)d_in[13];

    char* ws = (char*)d_ws;
    __bf16* lnbuf   = (__bf16*)(ws);                         //  8 MiB (reused)
    __bf16* qkvbuf  = (__bf16*)(ws + (8ull  << 20));         // 24 MiB
    __bf16* attnbuf = (__bf16*)(ws + (32ull << 20));         //  8 MiB
    float*  hbuf    = (float*)(ws + (40ull << 20));          // 16 MiB
    __bf16* ffn1buf = (__bf16*)(ws + (56ull << 20));         // 32 MiB

    const dim3 blk256(256), blk128(128);

    // 1) ln1(x) -> bf16
    ln_kernel<<<dim3(MROWS), blk256, 0, stream>>>(x, ln1_w, ln1_b, lnbuf);
    // 2) qkv = ln1 @ qkv_w + qkv_b  -> bf16 [4096, 3072]
    gemm_kernel<false, false, true><<<dim3(3072 / 128, MROWS / 128), blk256, 0, stream>>>(
        lnbuf, qkv_w, qkv_b, nullptr, qkvbuf, 1024, 3072);
    // 3) causal flash attention -> bf16 [4096, 1024]
    attn_kernel<<<dim3(SEQ_L / 64, 2 * NHEAD), blk128, 0, stream>>>(qkvbuf, attnbuf);
    // 4) h = x + attn @ proj_w + proj_b -> f32
    gemm_kernel<false, true, false><<<dim3(1024 / 128, MROWS / 128), blk256, 0, stream>>>(
        attnbuf, proj_w, proj_b, x, hbuf, 1024, 1024);
    // 5) ln2(h) -> bf16
    ln_kernel<<<dim3(MROWS), blk256, 0, stream>>>(hbuf, ln2_w, ln2_b, lnbuf);
    // 6) gelu(ln2 @ ffn_w1 + ffn_b1) -> bf16 [4096, 4096]
    gemm_kernel<true, false, true><<<dim3(4096 / 128, MROWS / 128), blk256, 0, stream>>>(
        lnbuf, ffn_w1, ffn_b1, nullptr, ffn1buf, 1024, 4096);
    // 7) out = h + ffn1 @ ffn_w2 + ffn_b2 -> f32 d_out
    gemm_kernel<false, true, false><<<dim3(1024 / 128, MROWS / 128), blk256, 0, stream>>>(
        ffn1buf, ffn_w2, ffn_b2, hbuf, (float*)d_out, 4096, 1024);
}
